// A2_42752104464871
// MI455X (gfx1250) — compile-verified, matrix operand order checked by reading, not run on hardware
//
#include <hip/hip_runtime.h>

// ---------------------------------------------------------------------------
// CDNA5 (gfx1250) WMMA types & helpers
// ---------------------------------------------------------------------------
typedef __attribute__((ext_vector_type(16))) __bf16 v16bf;
typedef __attribute__((ext_vector_type(8)))  __bf16 bf16x8;
typedef __attribute__((ext_vector_type(8)))  float  v8f;

// 16-bit A/B operand layout (ISA 7.12.2): lanes 0-15 hold row/col = lane,
// K = {half*8 .. half*8+7} in VGPR0-3 and K = {16+half*8 ..} in VGPR4-7,
// where half = lane[4].  Both chunks are 8 contiguous bf16 -> two b128 LDS reads.
__device__ __forceinline__ v16bf load_frag(const __bf16* base, int lane) {
    int half = (lane >> 4) & 1;
    bf16x8 lo = *(const bf16x8*)(base + half * 8);
    bf16x8 hi = *(const bf16x8*)(base + 16 + half * 8);
    v16bf r;
#pragma unroll
    for (int j = 0; j < 8; ++j) { r[j] = lo[j]; r[j + 8] = hi[j]; }
    return r;
}

__device__ __forceinline__ v8f wmma_bf16(v16bf a, v16bf b, v8f c) {
    return __builtin_amdgcn_wmma_f32_16x16x32_bf16(
        false, a, false, b, (short)0, c, false, false);
}

// problem sizes
#define BB    16
#define CC    512
#define NN    4096
#define II    128

// ---------------------------------------------------------------------------
// workspace layout (bytes)
// ---------------------------------------------------------------------------
constexpr size_t MAPB     = (size_t)BB * II * NN * 2;          // one bf16 map: 16.78 MB
constexpr size_t OFF_WCAT = 0;                                  // 384*512 bf16
constexpr size_t OFF_WPB  = OFF_WCAT + (size_t)384 * 512 * 2;   // 512*128 bf16
constexpr size_t OFF_A    = OFF_WPB + (size_t)512 * 128 * 2;
constexpr size_t OFF_B    = OFF_A + MAPB;
constexpr size_t OFF_V    = OFF_B + MAPB;
constexpr size_t OFF_Y    = OFF_V + MAPB;
constexpr size_t OFF_G    = OFF_Y + MAPB;                       // 16*128*128 f32
constexpr size_t OFF_M2   = OFF_G + (size_t)BB * II * II * 4;   // 128*128 f32
constexpr size_t OFF_SY   = OFF_M2 + (size_t)II * II * 4;       // 128 f32
constexpr size_t OFF_SC   = OFF_SY + 1024;                      // 512 f32
constexpr size_t OFF_SH   = OFF_SC + 2048;                      // 512 f32

// ---------------------------------------------------------------------------
// K0: convert weights to bf16, zero accumulators
// ---------------------------------------------------------------------------
__global__ void prep_kernel(const float* wA, const float* wB, const float* wV,
                            const float* wP, __bf16* wcat, __bf16* wPb,
                            float* G, float* M2) {
    int tid = blockIdx.x * blockDim.x + threadIdx.x;
    int stride = gridDim.x * blockDim.x;
    for (int idx = tid; idx < 384 * 512; idx += stride) {
        int blk = idx / (128 * 512), rem = idx % (128 * 512);
        const float* src = (blk == 0) ? wA : (blk == 1) ? wB : wV;
        wcat[idx] = (__bf16)src[rem];
    }
    for (int idx = tid; idx < 512 * 128; idx += stride) wPb[idx] = (__bf16)wP[idx];
    for (int idx = tid; idx < BB * II * II; idx += stride) G[idx] = 0.0f;
    for (int idx = tid; idx < II * II; idx += stride) M2[idx] = 0.0f;
}

// ---------------------------------------------------------------------------
// K1: projections.  blockIdx.x selects which 128-row weight block (A/B/V map);
// each WG computes one 128x128 output block of  W[128x512] * x_b[512x4096].
// 8 waves, wave w: rows 16w..16w+15, 8 n-tiles -> 8 acc tiles (64 VGPRs, no spill).
// grid (3, N/128, B)
// ---------------------------------------------------------------------------
__global__ __launch_bounds__(256, 1) void proj_gemm(
    const __bf16* __restrict__ wcat, const float* __restrict__ x,
    __bf16* __restrict__ Ab, __bf16* __restrict__ Bb, __bf16* __restrict__ Vb) {
    __shared__ __bf16 Wt[128 * 32];   // [row][k]
    __shared__ __bf16 Xt[128 * 32];   // [n][k]  (transposed)
    int tid = threadIdx.x, wv = tid >> 5, lane = tid & 31;
    int rc = lane & 15, half = (lane >> 4) & 1;
    int mapblk = blockIdx.x;              // 0:A 1:B 2:V
    int n0 = blockIdx.y * 128, b = blockIdx.z;
    const __bf16* W = wcat + (size_t)mapblk * 128 * 512;
    __bf16* dst = (mapblk == 0) ? Ab : (mapblk == 1) ? Bb : Vb;
    const float* X = x + (size_t)b * CC * NN;

    v8f acc[8];
#pragma unroll
    for (int nt = 0; nt < 8; ++nt)
#pragma unroll
        for (int e = 0; e < 8; ++e) acc[nt][e] = 0.0f;

    for (int ks = 0; ks < 16; ++ks) {
        int k0 = ks * 32;
        for (int idx = tid; idx < 128 * 4; idx += 256) {       // weight tile
            int row = idx >> 2, k8 = (idx & 3) * 8;
            *(bf16x8*)&Wt[row * 32 + k8] =
                *(const bf16x8*)&W[(size_t)row * 512 + k0 + k8];
        }
        for (int idx = tid; idx < 32 * 32; idx += 256) {       // x tile f32->bf16 transpose
            int kr = idx >> 5, nq = (idx & 31) * 4;
            const float* src = &X[(size_t)(k0 + kr) * NN + n0 + nq];
#pragma unroll
            for (int j = 0; j < 4; ++j) Xt[(nq + j) * 32 + kr] = (__bf16)src[j];
        }
        __syncthreads();
        v16bf af = load_frag(&Wt[(16 * wv + rc) * 32], lane);
#pragma unroll
        for (int nt = 0; nt < 8; ++nt) {
            v16bf bf = load_frag(&Xt[(nt * 16 + rc) * 32], lane);
            acc[nt] = wmma_bf16(af, bf, acc[nt]);
        }
        __syncthreads();
    }
#pragma unroll
    for (int nt = 0; nt < 8; ++nt)
#pragma unroll
        for (int vv = 0; vv < 8; ++vv) {
            int i = 16 * wv + vv + 8 * half, n = n0 + nt * 16 + rc;
            dst[((size_t)(b * II + i)) * NN + n] = (__bf16)acc[nt][vv];
        }
}

// ---------------------------------------------------------------------------
// K2: softmax over N for each (b,i) row of A_map, in place. grid = B*II
// ---------------------------------------------------------------------------
__global__ void softmax_rows(__bf16* Abuf) {
    __shared__ float red[256];
    int tid = threadIdx.x;
    __bf16* p = Abuf + (size_t)blockIdx.x * NN;
    float v[16], mx = -1e30f;
#pragma unroll
    for (int j = 0; j < 16; ++j) { v[j] = (float)p[tid + j * 256]; mx = fmaxf(mx, v[j]); }
    red[tid] = mx; __syncthreads();
    for (int s = 128; s > 0; s >>= 1) { if (tid < s) red[tid] = fmaxf(red[tid], red[tid + s]); __syncthreads(); }
    mx = red[0]; __syncthreads();
    float sum = 0.0f;
#pragma unroll
    for (int j = 0; j < 16; ++j) { v[j] = __expf(v[j] - mx); sum += v[j]; }
    red[tid] = sum; __syncthreads();
    for (int s = 128; s > 0; s >>= 1) { if (tid < s) red[tid] += red[tid + s]; __syncthreads(); }
    float inv = 1.0f / red[0];
#pragma unroll
    for (int j = 0; j < 16; ++j) p[tid + j * 256] = (__bf16)(v[j] * inv);
}

// ---------------------------------------------------------------------------
// K3: softmax over the 128 channels for each (b,n) column of V_map, in place.
// grid (N/64, B); tile [128 x 64] staged in LDS.
// ---------------------------------------------------------------------------
__global__ void softmax_cols(__bf16* Vbuf) {
    __shared__ __bf16 T[128 * 64];
    int tid = threadIdx.x, n0 = blockIdx.x * 64, b = blockIdx.y;
    for (int idx = tid; idx < 128 * 8; idx += 256) {
        int i = idx >> 3, n8 = (idx & 7) * 8;
        *(bf16x8*)&T[i * 64 + n8] =
            *(const bf16x8*)&Vbuf[((size_t)(b * II + i)) * NN + n0 + n8];
    }
    __syncthreads();
    if (tid < 64) {
        int col = tid;
        float mx = -1e30f;
        for (int i = 0; i < 128; ++i) mx = fmaxf(mx, (float)T[i * 64 + col]);
        float sum = 0.0f;
        for (int i = 0; i < 128; ++i) {
            float e = __expf((float)T[i * 64 + col] - mx);
            sum += e; T[i * 64 + col] = (__bf16)e;
        }
        float inv = 1.0f / sum;
        for (int i = 0; i < 128; ++i)
            T[i * 64 + col] = (__bf16)((float)T[i * 64 + col] * inv);
    }
    __syncthreads();
    for (int idx = tid; idx < 128 * 8; idx += 256) {
        int i = idx >> 3, n8 = (idx & 7) * 8;
        *(bf16x8*)&Vbuf[((size_t)(b * II + i)) * NN + n0 + n8] =
            *(const bf16x8*)&T[i * 64 + n8];
    }
}

// ---------------------------------------------------------------------------
// K4/K6: Out[128x128] (+)= A[128xN] * B[128xN]^T   (NT GEMM, K split over WGs,
// f32 atomic accumulation).  perBatch=1 -> per-batch output (G); 0 -> shared (M2).
// grid (B, 4 k-chunks), 8 waves, wave w: rows 16w, all 8 col-tiles.
// ---------------------------------------------------------------------------
__global__ __launch_bounds__(256, 1) void gemm_nt_acc(
    const __bf16* __restrict__ Abase, const __bf16* __restrict__ Bbase,
    float* Out, int perBatch) {
    __shared__ __bf16 At[128 * 32], Bt[128 * 32];
    int tid = threadIdx.x, wv = tid >> 5, lane = tid & 31;
    int rc = lane & 15, half = (lane >> 4) & 1;
    int b = blockIdx.x, kc = blockIdx.y;
    const __bf16* A = Abase + (size_t)b * II * NN;
    const __bf16* Bm = Bbase + (size_t)b * II * NN;
    float* O = Out + (perBatch ? (size_t)b * II * II : 0);

    v8f acc[8];
#pragma unroll
    for (int jt = 0; jt < 8; ++jt)
#pragma unroll
        for (int e = 0; e < 8; ++e) acc[jt][e] = 0.0f;

    for (int ks = 0; ks < 32; ++ks) {
        int k0 = kc * 1024 + ks * 32;
        for (int idx = tid; idx < 128 * 4; idx += 256) {
            int row = idx >> 2, k8 = (idx & 3) * 8;
            *(bf16x8*)&At[row * 32 + k8] = *(const bf16x8*)&A[(size_t)row * NN + k0 + k8];
            *(bf16x8*)&Bt[row * 32 + k8] = *(const bf16x8*)&Bm[(size_t)row * NN + k0 + k8];
        }
        __syncthreads();
        v16bf af = load_frag(&At[(16 * wv + rc) * 32], lane);
#pragma unroll
        for (int jt = 0; jt < 8; ++jt) {
            v16bf bf = load_frag(&Bt[(jt * 16 + rc) * 32], lane);
            acc[jt] = wmma_bf16(af, bf, acc[jt]);
        }
        __syncthreads();
    }
#pragma unroll
    for (int jt = 0; jt < 8; ++jt)
#pragma unroll
        for (int vv = 0; vv < 8; ++vv) {
            int mi = 16 * wv + vv + 8 * half, jj = jt * 16 + rc;
            unsafeAtomicAdd(&O[mi * II + jj], acc[jt][vv]);
        }
}

// ---------------------------------------------------------------------------
// K5: Y[128x4096] = G[128x128] * V_attn[128x4096]   (per batch, bf16 out)
// grid (N/128, B); wave w: rows 16w, 8 n-tiles.
// ---------------------------------------------------------------------------
__global__ __launch_bounds__(256, 1) void y_gemm(
    const float* __restrict__ Gbuf, const __bf16* __restrict__ Vattn,
    __bf16* __restrict__ Y) {
    __shared__ __bf16 Gt[128 * 32];   // [m][k]
    __shared__ __bf16 Vt[128 * 32];   // [n][k]  (transposed)
    int tid = threadIdx.x, wv = tid >> 5, lane = tid & 31;
    int rc = lane & 15, half = (lane >> 4) & 1;
    int n0 = blockIdx.x * 128, b = blockIdx.y;
    const float* G = Gbuf + (size_t)b * II * II;
    const __bf16* V = Vattn + (size_t)b * II * NN;
    __bf16* Yo = Y + (size_t)b * II * NN;

    v8f acc[8];
#pragma unroll
    for (int nt = 0; nt < 8; ++nt)
#pragma unroll
        for (int e = 0; e < 8; ++e) acc[nt][e] = 0.0f;

    for (int ks = 0; ks < 4; ++ks) {
        int k0 = ks * 32;
        for (int idx = tid; idx < 128 * 8; idx += 256) {       // G f32 -> bf16
            int row = idx >> 3, k4 = (idx & 7) * 4;
#pragma unroll
            for (int j = 0; j < 4; ++j)
                Gt[row * 32 + k4 + j] = (__bf16)G[row * II + k0 + k4 + j];
        }
        for (int idx = tid; idx < 32 * 16; idx += 256) {       // V tile transpose
            int kr = idx >> 4, n8 = (idx & 15) * 8;
            bf16x8 v = *(const bf16x8*)&V[(size_t)(k0 + kr) * NN + n0 + n8];
#pragma unroll
            for (int j = 0; j < 8; ++j) Vt[(n8 + j) * 32 + kr] = v[j];
        }
        __syncthreads();
        v16bf af = load_frag(&Gt[(16 * wv + rc) * 32], lane);
#pragma unroll
        for (int nt = 0; nt < 8; ++nt) {
            v16bf bf = load_frag(&Vt[(nt * 16 + rc) * 32], lane);
            acc[nt] = wmma_bf16(af, bf, acc[nt]);
        }
        __syncthreads();
    }
#pragma unroll
    for (int nt = 0; nt < 8; ++nt)
#pragma unroll
        for (int vv = 0; vv < 8; ++vv) {
            int mi = 16 * wv + vv + 8 * half, n = n0 + nt * 16 + rc;
            Yo[(size_t)mi * NN + n] = (__bf16)acc[nt][vv];
        }
}

// ---------------------------------------------------------------------------
// K6b: S_Y[i] = sum over (b,n) of Y[b][i][n].  grid = 128
// ---------------------------------------------------------------------------
__global__ void row_sums(const __bf16* __restrict__ Y, float* SY) {
    __shared__ float red[256];
    int tid = threadIdx.x, i = blockIdx.x;
    float s = 0.0f;
    for (int b = 0; b < BB; ++b) {
        const __bf16* p = Y + ((size_t)(b * II + i)) * NN;
        for (int j = tid; j < NN; j += 256) s += (float)p[j];
    }
    red[tid] = s; __syncthreads();
    for (int st = 128; st > 0; st >>= 1) { if (tid < st) red[tid] += red[tid + st]; __syncthreads(); }
    if (tid == 0) SY[i] = red[0];
}

// ---------------------------------------------------------------------------
// K7: per-channel BN stats from S_Y and M2:
//   mean_c = wP[c].S_Y/(BN);  E2_c = wP[c]^T M2 wP[c]/(BN);  var = E2 - mean^2
//   sc = gamma*rsqrt(var+eps); sh = beta - mean*sc.   One WG of 512 threads.
// ---------------------------------------------------------------------------
__global__ void bn_stats(const float* __restrict__ wP, const float* __restrict__ SY,
                         const float* __restrict__ M2, const float* __restrict__ gamma,
                         const float* __restrict__ beta, float* sc, float* sh) {
    int c = threadIdx.x;
    const float* wrow = wP + (size_t)c * II;
    const float invBN = 1.0f / ((float)BB * (float)NN);
    float mean = 0.0f;
    for (int i = 0; i < II; ++i) mean += wrow[i] * SY[i];
    mean *= invBN;
    float e2 = 0.0f;
    for (int i = 0; i < II; ++i) {
        float t = 0.0f;
        for (int j = 0; j < II; ++j) t += M2[i * II + j] * wrow[j];
        e2 += wrow[i] * t;
    }
    e2 *= invBN;
    float var = e2 - mean * mean;
    float is = rsqrtf(var + 1e-5f);
    float s = gamma[c] * is;
    sc[c] = s;
    sh[c] = beta[c] - mean * s;
}

// ---------------------------------------------------------------------------
// K8: out = x + (wP*Y)*sc + sh.   Z GEMM fused with normalize + residual.
// grid (N/128, B, C/128); wave w: rows mb+16w, 8 n-tiles.
// ---------------------------------------------------------------------------
__global__ __launch_bounds__(256, 1) void z_final(
    const __bf16* __restrict__ wPb, const __bf16* __restrict__ Y,
    const float* __restrict__ x, const float* __restrict__ sc,
    const float* __restrict__ sh, float* __restrict__ out) {
    __shared__ __bf16 Pt[128 * 32];   // [m][k]
    __shared__ __bf16 Yt[128 * 32];   // [n][k]  (transposed)
    int tid = threadIdx.x, wv = tid >> 5, lane = tid & 31;
    int rc = lane & 15, half = (lane >> 4) & 1;
    int n0 = blockIdx.x * 128, b = blockIdx.y, mb = blockIdx.z * 128;
    const __bf16* Yb = Y + (size_t)b * II * NN;

    v8f acc[8];
#pragma unroll
    for (int nt = 0; nt < 8; ++nt)
#pragma unroll
        for (int e = 0; e < 8; ++e) acc[nt][e] = 0.0f;

    for (int ks = 0; ks < 4; ++ks) {
        int k0 = ks * 32;
        for (int idx = tid; idx < 128 * 4; idx += 256) {
            int row = idx >> 2, k8 = (idx & 3) * 8;
            *(bf16x8*)&Pt[row * 32 + k8] =
                *(const bf16x8*)&wPb[(size_t)(mb + row) * II + k0 + k8];
        }
        for (int idx = tid; idx < 32 * 16; idx += 256) {
            int kr = idx >> 4, n8 = (idx & 15) * 8;
            bf16x8 v = *(const bf16x8*)&Yb[(size_t)(k0 + kr) * NN + n0 + n8];
#pragma unroll
            for (int j = 0; j < 8; ++j) Yt[(n8 + j) * 32 + kr] = v[j];
        }
        __syncthreads();
        v16bf af = load_frag(&Pt[(16 * wv + rc) * 32], lane);
#pragma unroll
        for (int nt = 0; nt < 8; ++nt) {
            v16bf bf = load_frag(&Yt[(nt * 16 + rc) * 32], lane);
            acc[nt] = wmma_bf16(af, bf, acc[nt]);
        }
        __syncthreads();
    }
#pragma unroll
    for (int nt = 0; nt < 8; ++nt)
#pragma unroll
        for (int vv = 0; vv < 8; ++vv) {
            int c = mb + 16 * wv + vv + 8 * half;
            int n = n0 + nt * 16 + rc;
            size_t idx = ((size_t)b * CC + c) * NN + n;
            out[idx] = x[idx] + acc[nt][vv] * sc[c] + sh[c];
        }
}

// ---------------------------------------------------------------------------
extern "C" void kernel_launch(void* const* d_in, const int* in_sizes, int n_in,
                              void* d_out, int out_size, void* d_ws, size_t ws_size,
                              hipStream_t stream) {
    const float* x     = (const float*)d_in[0];
    const float* wA    = (const float*)d_in[1];
    const float* wB    = (const float*)d_in[2];
    const float* wV    = (const float*)d_in[3];
    const float* wP    = (const float*)d_in[4];
    const float* gamma = (const float*)d_in[5];
    const float* beta  = (const float*)d_in[6];
    float* out = (float*)d_out;

    char* ws = (char*)d_ws;
    __bf16* wcat = (__bf16*)(ws + OFF_WCAT);
    __bf16* wPb  = (__bf16*)(ws + OFF_WPB);
    __bf16* Ab   = (__bf16*)(ws + OFF_A);
    __bf16* Bf   = (__bf16*)(ws + OFF_B);
    __bf16* Vb   = (__bf16*)(ws + OFF_V);
    __bf16* Y    = (__bf16*)(ws + OFF_Y);
    float*  G    = (float*)(ws + OFF_G);
    float*  M2   = (float*)(ws + OFF_M2);
    float*  SY   = (float*)(ws + OFF_SY);
    float*  sc   = (float*)(ws + OFF_SC);
    float*  sh   = (float*)(ws + OFF_SH);

    prep_kernel<<<dim3(512), 256, 0, stream>>>(wA, wB, wV, wP, wcat, wPb, G, M2);
    proj_gemm<<<dim3(3, NN / 128, BB), 256, 0, stream>>>(wcat, x, Ab, Bf, Vb);
    softmax_rows<<<dim3(BB * II), 256, 0, stream>>>(Ab);
    softmax_cols<<<dim3(NN / 64, BB), 256, 0, stream>>>(Vb);
    gemm_nt_acc<<<dim3(BB, 4), 256, 0, stream>>>(Bf, Ab, G, 1);        // G = B * A^T
    y_gemm<<<dim3(NN / 128, BB), 256, 0, stream>>>(G, Vb, Y);          // Y = G * V
    row_sums<<<dim3(II), 256, 0, stream>>>(Y, SY);
    gemm_nt_acc<<<dim3(BB, 4), 256, 0, stream>>>(Y, Y, M2, 0);         // M2 = sum Y Y^T
    bn_stats<<<dim3(1), 512, 0, stream>>>(wP, SY, M2, gamma, beta, sc, sh);
    z_final<<<dim3(NN / 128, BB, CC / 128), 256, 0, stream>>>(wPb, Y, x, sc, sh, out);
}
